// MultiHeadAttention_42649025249604
// MI455X (gfx1250) — matrix operand order repaired and checked by reading, not compiled, not run on hardware
//
#include <hip/hip_runtime.h>
#include <math.h>

typedef __attribute__((ext_vector_type(16))) __bf16 v16bf;
typedef __attribute__((ext_vector_type(8)))  float  v8f;

union Frag {
  v16bf v;
  uint4 u[2];
};

#define D_MODEL 1024
#define SEQ     2048
#define BATCH   4
#define NHEADS  16
#define HD      64
#define NQKV    3072
#define MROWS   (BATCH*SEQ)

__device__ __forceinline__ v8f wmma_bf16(v16bf a, v16bf b, v8f c) {
  return __builtin_amdgcn_wmma_f32_16x16x32_bf16(
      false, a, false, b, (short)0, c, false, false);
}

// CDNA5 async DMA: global -> LDS, 16B per lane, tracked by ASYNCcnt.
// LDS operand = wave-relative byte offset = low 32 bits of generic shared ptr.
__device__ __forceinline__ void async_copy_b128(__bf16* lds_dst, const __bf16* gsrc) {
  unsigned lds = (unsigned)(size_t)(void*)lds_dst;
  asm volatile("global_load_async_to_lds_b128 %0, %1, off"
               :: "v"(lds), "v"(gsrc)
               : "memory");
}

__device__ __forceinline__ void wait_asynccnt0() {
  asm volatile("s_wait_asynccnt 0" ::: "memory");
}

// ---------------- prep kernels ----------------

__global__ __launch_bounds__(256)
void cvt_f32_bf16(const float* __restrict__ in, __bf16* __restrict__ out, int n) {
  int i = blockIdx.x * 256 + threadIdx.x;
  if (i < n) out[i] = (__bf16)in[i];
}

// W_eff[n,k] = W_qkv[n,k] + (alpha/r) * sum_r lora_B[n,r] * lora_A[r,k]
__global__ __launch_bounds__(256)
void build_weff(const float* __restrict__ Wq, const float* __restrict__ lA,
                const float* __restrict__ lB, __bf16* __restrict__ we) {
  int i = blockIdx.x * 256 + threadIdx.x;          // [0, 3072*1024)
  int nrow = i >> 10, kk = i & 1023;
  float acc = Wq[i];
  #pragma unroll
  for (int r = 0; r < 8; ++r)
    acc = fmaf(0.125f * lB[nrow * 8 + r], lA[r * 1024 + kk], acc);
  we[i] = (__bf16)acc;
}

// ---------------- WMMA GEMM: C[M,N] = A[M,K] * W[N,K]^T + bias ----------------
// BM=128, BN=128, BK=32; 8 waves in 4x2; wave tile 32x64 = 2x4 WMMA C tiles.
// Double-buffered LDS tiles filled by async-to-LDS DMA, overlapped with WMMA.
// MODE 0: fp32 store to Cout
// MODE 1: qkv epilogue -> bf16 q[B,H,S,Hd] (pre-scaled by 1/8), k[B,H,S,Hd], vT[B,H,Hd,S]

template<int MODE>
__global__ __launch_bounds__(256)
void wmma_gemm_bf16(const __bf16* __restrict__ A,
                    const __bf16* __restrict__ W,
                    const float* __restrict__ bias,
                    float* __restrict__ Cout,
                    __bf16* __restrict__ qb,
                    __bf16* __restrict__ kb,
                    __bf16* __restrict__ vtb,
                    int K, int N)
{
  constexpr int BM = 128, BN = 128, BKc = 32;
  constexpr int ASTR = 40, BSTR = 40;                  // bf16 elems; keeps 16B alignment
  __shared__ alignas(16) __bf16 As[2][BM * ASTR];
  __shared__ alignas(16) __bf16 Bs[2][BN * BSTR];

  const int tid  = threadIdx.x;
  const int wave = tid >> 5;
  const int lane = tid & 31;
  const int ll   = lane & 15;
  const int lh   = lane >> 4;
  const int m0 = blockIdx.y * BM;
  const int n0 = blockIdx.x * BN;
  const int wm = (wave >> 1) * 32;                     // 4 waves along M (32 rows each)
  const int wn = (wave & 1) * 64;                      // 2 waves along N (64 cols each)

  // per-thread staging assignment (4 async b128 per tile: 2 for A, 2 for B)
  const int ar0 = tid >> 2,           ac0 = (tid & 3) * 8;
  const int ar1 = (tid + 256) >> 2,   ac1 = ((tid + 256) & 3) * 8;
  const int br0 = ar0,                bc0 = ac0;
  const int br1 = ar1,                bc1 = ac1;

  v8f acc[2][4] = {};

  // prologue: stage tile 0 into buffer 0
  async_copy_b128(&As[0][ar0 * ASTR + ac0], &A[(size_t)(m0 + ar0) * K + ac0]);
  async_copy_b128(&As[0][ar1 * ASTR + ac1], &A[(size_t)(m0 + ar1) * K + ac1]);
  async_copy_b128(&Bs[0][br0 * BSTR + bc0], &W[(size_t)(n0 + br0) * K + bc0]);
  async_copy_b128(&Bs[0][br1 * BSTR + bc1], &W[(size_t)(n0 + br1) * K + bc1]);

  int cur = 0;
  for (int k0 = 0; k0 < K; k0 += BKc) {
    wait_asynccnt0();
    __syncthreads();                                   // tile k0 fully in LDS

    Frag af[2], bw[4];
    #pragma unroll
    for (int mt = 0; mt < 2; ++mt) {
      const __bf16* p = &As[cur][(wm + mt * 16 + ll) * ASTR];
      af[mt].u[0] = *(const uint4*)(p + lh * 8);         // K = lh*8 .. +7
      af[mt].u[1] = *(const uint4*)(p + 16 + lh * 8);    // K = 16+lh*8 .. +7
    }
    #pragma unroll
    for (int nt = 0; nt < 4; ++nt) {
      const __bf16* p = &Bs[cur][(wn + nt * 16 + ll) * BSTR + lh * 8];
      bw[nt].u[0] = *(const uint4*)(p);                  // K = lh*8 .. +7
      bw[nt].u[1] = *(const uint4*)(p + 16);             // K = 16+lh*8 .. +7
    }
    __syncthreads();                                   // all frag reads retired

    // stage next tile into the other buffer; DMA overlaps the WMMAs below
    {
      int kn = (k0 + BKc < K) ? (k0 + BKc) : k0;
      int nb = cur ^ 1;
      async_copy_b128(&As[nb][ar0 * ASTR + ac0], &A[(size_t)(m0 + ar0) * K + kn + ac0]);
      async_copy_b128(&As[nb][ar1 * ASTR + ac1], &A[(size_t)(m0 + ar1) * K + kn + ac1]);
      async_copy_b128(&Bs[nb][br0 * BSTR + bc0], &W[(size_t)(n0 + br0) * K + kn + bc0]);
      async_copy_b128(&Bs[nb][br1 * BSTR + bc1], &W[(size_t)(n0 + br1) * K + kn + bc1]);
    }

    #pragma unroll
    for (int mt = 0; mt < 2; ++mt)
      #pragma unroll
      for (int nt = 0; nt < 4; ++nt)
        acc[mt][nt] = wmma_bf16(af[mt].v, bw[nt].v, acc[mt][nt]);

    cur ^= 1;
  }

  // epilogue: C layout -> row = j + 8*lh, col = ll within 16x16 tile
  #pragma unroll
  for (int mt = 0; mt < 2; ++mt) {
    #pragma unroll
    for (int nt = 0; nt < 4; ++nt) {
      #pragma unroll
      for (int j = 0; j < 8; ++j) {
        int m = m0 + wm + mt * 16 + j + lh * 8;
        int n = n0 + wn + nt * 16 + ll;
        float val = acc[mt][nt][j] + bias[n];
        if (MODE == 0) {
          Cout[(size_t)m * N + n] = val;
        } else {
          int t = n >> 10;            // 0=q 1=k 2=v
          int h = (n >> 6) & 15;
          int d = n & 63;
          int b = m >> 11;
          int s = m & 2047;
          size_t bh = (size_t)(b * NHEADS + h);
          if (t == 0) {
            qb[(bh * SEQ + s) * HD + d] = (__bf16)(val * 0.125f);  // fold softmax scale
          } else if (t == 1) {
            kb[(bh * SEQ + s) * HD + d] = (__bf16)val;
          } else {
            vtb[(bh * HD + d) * SEQ + s] = (__bf16)val;            // transposed V
          }
        }
      }
    }
  }
}

// ---------------- flash attention ----------------
// One block = one (b,h) x 128 queries (8 waves x 16 rows).
// K/V chunks (32 keys) async-staged into double-buffered LDS shared by all waves.
// Softmax denominator accumulated with a WMMA against an all-ones B matrix
// (row sums of P land replicated per lane in C layout -> no shuffle reduction).

__global__ __launch_bounds__(256)
void flash_attn(const __bf16* __restrict__ q,
                const __bf16* __restrict__ k,
                const __bf16* __restrict__ vt,
                __bf16* __restrict__ ctx)
{
  constexpr int KSTR = 72;   // K tile row stride (64 hd + pad), 16B aligned
  constexpr int VSTR = 40;   // V tile row stride (32 keys + pad)
  __shared__ alignas(16) __bf16 kbuf[2][32 * KSTR];
  __shared__ alignas(16) __bf16 vbuf[2][64 * VSTR];
  __shared__ alignas(16) __bf16 pls[8 * 16 * 40];    // per-wave 16x32 P tile

  const int tid  = threadIdx.x;
  const int wave = tid >> 5;
  const int lane = tid & 31;
  const int ll   = lane & 15;
  const int lh   = lane >> 4;
  const int bh   = blockIdx.x >> 4;                  // 0..63
  const int qblk = blockIdx.x & 15;
  const int q0   = qblk * 128 + wave * 16;

  const __bf16* qp = q  + (size_t)bh * SEQ * HD;
  const __bf16* kp = k  + (size_t)bh * SEQ * HD;
  const __bf16* vp = vt + (size_t)bh * HD * SEQ;
  __bf16* pl = &pls[wave * 16 * 40];

  // staging assignment: 256 chunks K (32 rows x 8) + 256 chunks V (64 rows x 4)
  const int kr = tid >> 3, kc = (tid & 7) * 8;
  const int vr = tid >> 2, vc = (tid & 3) * 8;

  // Q fragments: A-layout, rows q0..q0+15, Hd split 0-31 / 32-63 (q pre-scaled)
  Frag qf[2];
  {
    const __bf16* r = qp + (size_t)(q0 + ll) * HD;
    qf[0].u[0] = *(const uint4*)(r + lh * 8);
    qf[0].u[1] = *(const uint4*)(r + 16 + lh * 8);
    qf[1].u[0] = *(const uint4*)(r + 32 + lh * 8);
    qf[1].u[1] = *(const uint4*)(r + 48 + lh * 8);
  }

  // constant all-ones B fragment (bf16 1.0 = 0x3F80)
  Frag ones;
  ones.u[0] = make_uint4(0x3F803F80u, 0x3F803F80u, 0x3F803F80u, 0x3F803F80u);
  ones.u[1] = ones.u[0];

  v8f o[4] = {};
  v8f sumacc = {};                                   // running softmax denominator (per row)
  float mrun[8];
  #pragma unroll
  for (int j = 0; j < 8; ++j) mrun[j] = -1e30f;

  // prologue: stage key chunk 0 into buffer 0
  async_copy_b128(&kbuf[0][kr * KSTR + kc], kp + (size_t)kr * HD + kc);
  async_copy_b128(&vbuf[0][vr * VSTR + vc], vp + (size_t)vr * SEQ + vc);

  int cur = 0;
  for (int kj = 0; kj < SEQ; kj += 32) {
    wait_asynccnt0();
    __syncthreads();                                 // chunk kj staged

    // K fragments: B-layout, col = key, inner = Hd slice
    Frag kf[2][2];
    #pragma unroll
    for (int t = 0; t < 2; ++t)
      #pragma unroll
      for (int is = 0; is < 2; ++is) {
        const __bf16* r = &kbuf[cur][(t * 16 + ll) * KSTR + is * 32 + lh * 8];
        kf[t][is].u[0] = *(const uint4*)(r);
        kf[t][is].u[1] = *(const uint4*)(r + 16);
      }
    __syncthreads();                                 // K frag reads retired (V read later, fenced
                                                     // by next iteration's top barrier)
    // stage next chunk; DMA overlaps all math below
    {
      int kn = (kj + 32 < SEQ) ? (kj + 32) : kj;
      int nb = cur ^ 1;
      async_copy_b128(&kbuf[nb][kr * KSTR + kc], kp + (size_t)(kn + kr) * HD + kc);
      async_copy_b128(&vbuf[nb][vr * VSTR + vc], vp + (size_t)vr * SEQ + kn + vc);
    }

    v8f s0 = {}, s1 = {};
    s0 = wmma_bf16(qf[0].v, kf[0][0].v, s0);
    s0 = wmma_bf16(qf[1].v, kf[0][1].v, s0);
    s1 = wmma_bf16(qf[0].v, kf[1][0].v, s1);
    s1 = wmma_bf16(qf[1].v, kf[1][1].v, s1);

    // online softmax max over 32 key columns; row r = j + 8*lh lives in one 16-lane half
    float p0[8], p1[8], alpha[8];
    #pragma unroll
    for (int j = 0; j < 8; ++j) {
      float a  = s0[j];
      float b2 = s1[j];
      float mx = fmaxf(a, b2);
      #pragma unroll
      for (int d2 = 1; d2 < 16; d2 <<= 1) mx = fmaxf(mx, __shfl_xor(mx, d2));
      float mn = fmaxf(mrun[j], mx);
      alpha[j] = __expf(mrun[j] - mn);
      mrun[j] = mn;
      p0[j] = __expf(a - mn);
      p1[j] = __expf(b2 - mn);
    }

    // rescale O and denominator, write P (C-layout) to this wave's LDS tile as bf16
    #pragma unroll
    for (int j = 0; j < 8; ++j) {
      o[0][j] *= alpha[j]; o[1][j] *= alpha[j];
      o[2][j] *= alpha[j]; o[3][j] *= alpha[j];
      sumacc[j] *= alpha[j];
      int row = j + lh * 8;
      pl[row * 40 + ll]      = (__bf16)p0[j];
      pl[row * 40 + 16 + ll] = (__bf16)p1[j];
    }
    // cross-lane LDS dependency within the wave
    asm volatile("s_wait_dscnt 0" ::: "memory");

    // reload P in A-layout (16 x 32 keys)
    Frag pf;
    {
      const __bf16* r = &pl[ll * 40];
      pf.u[0] = *(const uint4*)(r + lh * 8);
      pf.u[1] = *(const uint4*)(r + 16 + lh * 8);
    }
    asm volatile("" ::: "memory");

    // denominator: row sums of P via WMMA against ones (replicated across columns)
    sumacc = wmma_bf16(pf.v, ones.v, sumacc);

    // V fragments: B-layout from transposed V tile [hd][keys]; O += P * V
    #pragma unroll
    for (int t = 0; t < 4; ++t) {
      const __bf16* r = &vbuf[cur][(t * 16 + ll) * VSTR + lh * 8];
      Frag vf;
      vf.u[0] = *(const uint4*)(r);
      vf.u[1] = *(const uint4*)(r + 16);
      o[t] = wmma_bf16(pf.v, vf.v, o[t]);
    }

    cur ^= 1;
  }

  // finalize: divide by denominator, write ctx [B*S, D] bf16
  const int b = bh >> 4, h = bh & 15;
  #pragma unroll
  for (int j = 0; j < 8; ++j) {
    float inv = 1.0f / sumacc[j];
    int row = q0 + j + lh * 8;
    size_t base = ((size_t)(b * SEQ + row)) * D_MODEL + h * HD + ll;
    ctx[base + 0]  = (__bf16)(o[0][j] * inv);
    ctx[base + 16] = (__bf16)(o[1][j] * inv);
    ctx[base + 32] = (__bf16)(o[2][j] * inv);
    ctx[base + 48] = (__bf16)(o[3][j] * inv);
  }
}

// ---------------- launcher ----------------

extern "C" void kernel_launch(void* const* d_in, const int* in_sizes, int n_in,
                              void* d_out, int out_size, void* d_ws, size_t ws_size,
                              hipStream_t stream)
{
  const float* x     = (const float*)d_in[0];
  const float* lA    = (const float*)d_in[1];
  const float* lB    = (const float*)d_in[2];
  const float* Wqkv  = (const float*)d_in[3];
  const float* bqkv  = (const float*)d_in[4];
  const float* Wproj = (const float*)d_in[5];
  const float* bproj = (const float*)d_in[6];
  float* out = (float*)d_out;

  char* ws = (char*)d_ws;
  size_t off = 0;
  auto alloc = [&](size_t bytes) -> void* {
    void* p = ws + off;
    off += (bytes + 255) & ~(size_t)255;
    return p;
  };

  __bf16* xb   = (__bf16*)alloc((size_t)MROWS * D_MODEL * 2);
  __bf16* weff = (__bf16*)alloc((size_t)NQKV * D_MODEL * 2);
  __bf16* wpb  = (__bf16*)alloc((size_t)D_MODEL * D_MODEL * 2);
  __bf16* qb   = (__bf16*)alloc((size_t)BATCH * NHEADS * SEQ * HD * 2);
  __bf16* kb   = (__bf16*)alloc((size_t)BATCH * NHEADS * SEQ * HD * 2);
  __bf16* vtb  = (__bf16*)alloc((size_t)BATCH * NHEADS * SEQ * HD * 2);
  __bf16* ctx  = (__bf16*)alloc((size_t)MROWS * D_MODEL * 2);
  (void)ws_size; (void)in_sizes; (void)n_in; (void)out_size;

  // prep: downconvert x / W_proj, fold LoRA into effective QKV weight
  cvt_f32_bf16<<<(MROWS * D_MODEL) / 256, 256, 0, stream>>>(x, xb, MROWS * D_MODEL);
  cvt_f32_bf16<<<(D_MODEL * D_MODEL) / 256, 256, 0, stream>>>(Wproj, wpb, D_MODEL * D_MODEL);
  build_weff<<<(NQKV * D_MODEL) / 256, 256, 0, stream>>>(Wqkv, lA, lB, weff);

  // fused QKV(+LoRA) GEMM -> q (pre-scaled), k, vT (bf16)
  dim3 g1(NQKV / 128, MROWS / 128);
  wmma_gemm_bf16<1><<<g1, 256, 0, stream>>>(xb, weff, bqkv, nullptr,
                                            qb, kb, vtb, D_MODEL, NQKV);

  // flash attention: 64 (b,h) x 16 query-blocks
  flash_attn<<<BATCH * NHEADS * (SEQ / 128), 256, 0, stream>>>(qb, kb, vtb, ctx);

  // output projection -> fp32 out
  dim3 g2(D_MODEL / 128, MROWS / 128);
  wmma_gemm_bf16<0><<<g2, 256, 0, stream>>>(ctx, wpb, bproj, out,
                                            nullptr, nullptr, nullptr, D_MODEL, D_MODEL);
}
